// SAPIAttention_13451837571663
// MI455X (gfx1250) — matrix operand order repaired and checked by reading, compiled, not run on hardware
//
#include <hip/hip_runtime.h>
#include <hip/hip_bf16.h>
#include <stdint.h>

// Problem constants (match reference)
#define B_    2
#define S_    2048
#define HID_  2048
#define H_    32
#define KVH_  8
#define D_    64
#define G_    (H_ / KVH_)
#define ROT_  32
#define NEGV  (-1000000000.0f)

typedef __attribute__((ext_vector_type(16))) __bf16 bf16x16;
typedef __attribute__((ext_vector_type(8)))  float  v8f;
typedef __attribute__((ext_vector_type(4)))  unsigned int u32x4;
typedef __attribute__((ext_vector_type(8)))  unsigned int u32x8;

// ---------- helpers ----------
static __device__ __forceinline__ unsigned short f2bf(float f) {
  unsigned int u = __builtin_bit_cast(unsigned int, f);
  u += 0x7fffu + ((u >> 16) & 1u);           // round-to-nearest-even
  return (unsigned short)(u >> 16);
}
static __device__ __forceinline__ float bf2f(unsigned short h) {
  unsigned int u = ((unsigned int)h) << 16;
  return __builtin_bit_cast(float, u);
}
// Generic pointer -> hardware LDS byte address (ISA 10.2: LDS aperture uses addr[31:0])
static __device__ __forceinline__ unsigned lds_addr(const void* p) {
  return (unsigned)(uintptr_t)p;
}
// XOR-lane shuffle via ds_swizzle immediate (group-of-32 mode):
// offset = {xor_mask[14:10], or_mask[9:5]=0, and_mask[4:0]=0x1f}.
// No address VALU, no bounds check (vs ds_bpermute lowering of __shfl_xor).
template <int MASK>
static __device__ __forceinline__ float swz_xor(float x) {
  int i = __builtin_bit_cast(int, x);
  i = __builtin_amdgcn_ds_swizzle(i, (MASK << 10) | 0x1f);
  return __builtin_bit_cast(float, i);
}
// Reduce across the 16-lane half (xor bits 0..3 never cross the half boundary)
static __device__ __forceinline__ float half16_max(float x) {
  x = fmaxf(x, swz_xor<1>(x));
  x = fmaxf(x, swz_xor<2>(x));
  x = fmaxf(x, swz_xor<4>(x));
  x = fmaxf(x, swz_xor<8>(x));
  return x;
}
static __device__ __forceinline__ float half16_sum(float x) {
  x += swz_xor<1>(x);
  x += swz_xor<2>(x);
  x += swz_xor<4>(x);
  x += swz_xor<8>(x);
  return x;
}
// Async DMA: global -> LDS, 16B per lane, tracked by ASYNCcnt.
// INST_OFFSET is added to BOTH the LDS and global addresses (ISA 10, async ops).
#define ASYNC_CP_B128(ldsoff, gptr, imm)                                      \
  asm volatile("global_load_async_to_lds_b128 %0, %1, off offset:" #imm       \
               :: "v"(ldsoff), "v"(gptr) : "memory")
#define ASYNC_WAIT0() asm volatile("s_wait_asynccnt 0x0" ::: "memory")

struct Frag32 { uint4 lo, hi; };

// A-matrix 16x32 bf16 fragment (ISA 7.12.2): lanes 0-15 hold K={0..7,16..23},
// lanes 16-31 hold K={8..15,24..31}; M = lane&15 selects the row pointer.
static __device__ __forceinline__ bf16x16 ld_afrag(const unsigned short* rowp, int lane) {
  const int kh = (lane >> 4) * 8;
  Frag32 f;
  f.lo = *reinterpret_cast<const uint4*>(rowp + kh);
  f.hi = *reinterpret_cast<const uint4*>(rowp + 16 + kh);
  return __builtin_bit_cast(bf16x16, f);
}
// B-matrix 32x16 bf16 fragment: lanes 0-15 hold K={0..7,16..23}, lanes 16-31 K={8..15,24..31},
// N = lane&15 selects the row pointer (row-major [N][K] staging).
static __device__ __forceinline__ bf16x16 ld_bfrag(const unsigned short* rowp, int lane) {
  const int kh = (lane >> 4) * 8;
  Frag32 f;
  f.lo = *reinterpret_cast<const uint4*>(rowp + kh);
  f.hi = *reinterpret_cast<const uint4*>(rowp + 16 + kh);
  return __builtin_bit_cast(bf16x16, f);
}
static __device__ __forceinline__ v8f wmma_bf16(bf16x16 a, bf16x16 b, v8f c) {
  return __builtin_amdgcn_wmma_f32_16x16x32_bf16(false, a, false, b, (short)0, c, false, false);
}

// ---------- f32 -> bf16 conversion ----------
__global__ void cvt_f32_bf16_kernel(const float* __restrict__ in,
                                    unsigned short* __restrict__ out, int n) {
  int i = blockIdx.x * blockDim.x + threadIdx.x;
  if (i < n) out[i] = f2bf(in[i]);
}

// ---------- tiled bf16 GEMM: C[M,N] = cscale * (A[M,K] * B[N,K]^T) ----------
// Double-buffered LDS tiles filled by async global->LDS DMA; the DMA for tile
// k+1 overlaps the WMMA work on tile k. One barrier + one asynccnt wait per step.
template <bool OUT_F32>
__global__ __launch_bounds__(256) void gemm_bf16_kernel(
    const unsigned short* __restrict__ A,   // M x K bf16
    const unsigned short* __restrict__ Bw,  // N x K bf16
    void* __restrict__ Cout, int M, int N, int K, float cscale) {
  __shared__ __align__(32) unsigned short As[2][128][32];
  __shared__ __align__(32) unsigned short Bs[2][128][32];

  const int tid  = threadIdx.x;
  const int lane = tid & 31;
  const int wave = tid >> 5;
  const int wm = wave & 3;   // 4 wave-strips of 32 rows (M)
  const int wn = wave >> 2;  // 2 wave-strips of 64 cols (N)
  const int m0 = blockIdx.y * 128;
  const int n0 = blockIdx.x * 128;
  const int ln = lane & 15;

  v8f acc[2][4];
#pragma unroll
  for (int i = 0; i < 2; i++)
#pragma unroll
    for (int j = 0; j < 4; j++) acc[i][j] = 0.f;

  const int ldr = tid >> 1;        // 0..127
  const int ldc = (tid & 1) * 16;  // 0 or 16
  const unsigned short* gA = A  + (size_t)(m0 + ldr) * K + ldc;
  const unsigned short* gB = Bw + (size_t)(n0 + ldr) * K + ldc;

  auto issue = [&](int k0, int buf) {
    const unsigned ao = lds_addr(&As[buf][ldr][ldc]);
    const unsigned bo = lds_addr(&Bs[buf][ldr][ldc]);
    const unsigned short* pa = gA + k0;
    const unsigned short* pb = gB + k0;
    ASYNC_CP_B128(ao, pa, 0);
    ASYNC_CP_B128(ao, pa, 16);
    ASYNC_CP_B128(bo, pb, 0);
    ASYNC_CP_B128(bo, pb, 16);
  };

  issue(0, 0);
  for (int k0 = 0; k0 < K; k0 += 32) {
    const int buf = (k0 >> 5) & 1;
    ASYNC_WAIT0();       // own DMA portion landed
    __syncthreads();     // everyone's portion landed; prev reads of buf^1 done
    if (k0 + 32 < K) issue(k0 + 32, buf ^ 1);  // prefetch next tile

    bf16x16 af[2];
#pragma unroll
    for (int mt = 0; mt < 2; mt++)
      af[mt] = ld_afrag(&As[buf][wm * 32 + mt * 16 + ln][0], lane);
#pragma unroll
    for (int nt = 0; nt < 4; nt++) {
      bf16x16 bfv = ld_bfrag(&Bs[buf][wn * 64 + nt * 16 + ln][0], lane);
#pragma unroll
      for (int mt = 0; mt < 2; mt++)
        acc[mt][nt] = wmma_bf16(af[mt], bfv, acc[mt][nt]);
    }
  }

  const int rhi = (lane >> 4) * 8;  // C-layout: M = vi + 8*(lane>=16)
#pragma unroll
  for (int mt = 0; mt < 2; mt++)
#pragma unroll
    for (int nt = 0; nt < 4; nt++)
#pragma unroll
      for (int vi = 0; vi < 8; vi++) {
        int m = m0 + wm * 32 + mt * 16 + vi + rhi;
        int n = n0 + wn * 64 + nt * 16 + ln;
        float v = acc[mt][nt][vi] * cscale;
        if (OUT_F32)
          reinterpret_cast<float*>(Cout)[(size_t)m * N + n] = v;
        else
          reinterpret_cast<unsigned short*>(Cout)[(size_t)m * N + n] = f2bf(v);
      }
}

// ---------- partial RoPE in place on [B,S,nh,D] bf16 ----------
__global__ void rope_kernel(unsigned short* __restrict__ X,
                            const float* __restrict__ cosb,
                            const float* __restrict__ sinb, int nh, int total) {
  int idx = blockIdx.x * blockDim.x + threadIdx.x;
  if (idx >= total) return;
  int i  = idx & 15;        // pair index within ROT/2
  int t  = idx >> 4;
  int h  = t % nh;
  int bs = t / nh;          // b*S + s
  unsigned short* p = X + ((size_t)bs * nh + h) * D_;
  const float* cb = cosb + (size_t)bs * ROT_;
  const float* sb = sinb + (size_t)bs * ROT_;
  float x0 = bf2f(p[i]);
  float x1 = bf2f(p[i + 16]);
  float o0 = x0 * cb[i]      - x1 * sb[i];       // rotate_half: -x[hi] on lo
  float o1 = x1 * cb[i + 16] + x0 * sb[i + 16];  //  x[lo] on hi
  p[i]      = f2bf(o0);
  p[i + 16] = f2bf(o1);
}

// ---------- flash attention (causal, GQA) ----------
// Q: [B,S,H,D] bf16 (rope applied, pre-scaled by 1/sqrt(D)),
// K/V: [B,S,KVH,D] bf16, O: [B,S,H,D] bf16.
__global__ __launch_bounds__(128) void flash_kernel(
    const unsigned short* __restrict__ Q,
    const unsigned short* __restrict__ Kb,
    const unsigned short* __restrict__ Vb,
    unsigned short* __restrict__ O) {
  __shared__ __align__(32) unsigned short Ks[64][64];      // [key][d]
  __shared__ __align__(32) unsigned short Vt[64][64];      // [d][key]
  __shared__ __align__(32) unsigned short Ps[4][16][64];   // per-wave P tile

  const int tid  = threadIdx.x;
  const int lane = tid & 31;
  const int wave = tid >> 5;
  const int qt  = blockIdx.x;
  const int h   = blockIdx.y;
  const int b   = blockIdx.z;
  const int kvh = h / G_;
  const int q0  = qt * 64 + wave * 16;   // this wave's 16 query rows
  const int ln  = lane & 15;
  const int rhi = (lane >> 4) * 8;

  // Persistent Q A-fragments (16 rows x D=64 -> two 16x32 frags)
  const unsigned short* qrow = Q + (((size_t)b * S_ + (q0 + ln)) * H_ + h) * D_;
  bf16x16 aq0 = ld_afrag(qrow, lane);
  bf16x16 aq1 = ld_afrag(qrow + 32, lane);

  v8f o[4];
#pragma unroll
  for (int i = 0; i < 4; i++) o[i] = 0.f;
  float mrun[8], lrun[8];
#pragma unroll
  for (int i = 0; i < 8; i++) { mrun[i] = -3.0e38f; lrun[i] = 0.f; }

  const int trow = tid >> 1;        // 0..63 key row
  const int tcol = (tid & 1) * 32;  // 0 or 32

  auto process_tile = [&](int kt, bool diag) {
    const int key0 = kt * 64;
    // --- TDM: one wave DMAs the strided 64x64 bf16 K tile into LDS ---
    if (wave == 0) {
      const unsigned long long ga = (unsigned long long)(uintptr_t)(
          Kb + (((size_t)b * S_ + key0) * KVH_ + kvh) * D_);
      u32x4 g0;
      g0[0] = 1u;                    // count=1 (valid user descriptor)
      g0[1] = lds_addr(&Ks[0][0]);   // LDS destination (bytes)
      g0[2] = (unsigned)ga;          // global_addr[31:0]
      g0[3] = ((unsigned)(ga >> 32) & 0x01ffffffu) | 0x80000000u;  // [56:32] | type=2
      u32x8 g1;
      g1[0] = 1u << 16;              // data_size=2B; workgroup_mask=0 (no cluster)
      g1[1] = 64u << 16;             // tensor_dim0 = 64 (bits 79:48, lo half)
      g1[2] = 64u << 16;             // tensor_dim0 hi=0 | tensor_dim1 lo = 64
      g1[3] = 64u << 16;             // tensor_dim1 hi=0 | tile_dim0 = 64
      g1[4] = 64u;                   // tile_dim1 = 64; tile_dim2 = 0 (2D tile)
      g1[5] = (unsigned)(KVH_ * D_); // tensor_dim0_stride = 512 elements
      g1[6] = 0u;
      g1[7] = 0u;
      u32x4 gz = {0u, 0u, 0u, 0u};   // groups 2/3: dims 3/4 unused (tile_dim=0)
      asm volatile("tensor_load_to_lds %0, %1, %2, %3"
                   :: "s"(g0), "s"(g1), "s"(gz), "s"(gz) : "memory");
    }
    // V tile: needs element transpose -> load + ds_store_b16 path (all waves)
    {
      const unsigned short* vp =
          Vb + (((size_t)b * S_ + (key0 + trow)) * KVH_ + kvh) * D_ + tcol;
#pragma unroll
      for (int j = 0; j < 32; j++) Vt[tcol + j][trow] = vp[j];
    }
    if (wave == 0) __builtin_amdgcn_s_wait_tensorcnt(0);  // K DMA landed
    __syncthreads();

    // S = Q * K^T  (16x64 scores per wave; K-dim D=64 as two chained WMMAs)
    v8f s[4];
#pragma unroll
    for (int nt = 0; nt < 4; nt++) {
      bf16x16 bk0 = ld_bfrag(&Ks[nt * 16 + ln][0], lane);
      bf16x16 bk1 = ld_bfrag(&Ks[nt * 16 + ln][32], lane);
      v8f c = 0.f;
      c = wmma_bf16(aq0, bk0, c);
      c = wmma_bf16(aq1, bk1, c);
      s[nt] = c;
    }

    // Causal mask: only the peeled diagonal tile pays for it.
    if (diag) {
#pragma unroll
      for (int nt = 0; nt < 4; nt++)
#pragma unroll
        for (int vi = 0; vi < 8; vi++)
          if ((key0 + nt * 16 + ln) > (q0 + vi + rhi)) s[nt][vi] += NEGV;
    }

    // Online softmax: per-row reductions across 16 lanes (ds_swizzle immediates)
#pragma unroll
    for (int vi = 0; vi < 8; vi++) {
      float mx = fmaxf(fmaxf(s[0][vi], s[1][vi]), fmaxf(s[2][vi], s[3][vi]));
      mx = half16_max(mx);
      float mnew  = fmaxf(mrun[vi], mx);
      float alpha = __expf(mrun[vi] - mnew);
      mrun[vi] = mnew;
      lrun[vi] *= alpha;
      float rs = 0.f;
#pragma unroll
      for (int nt = 0; nt < 4; nt++) {
        float p = __expf(s[nt][vi] - mnew);
        s[nt][vi] = p;
        rs += p;
      }
      lrun[vi] += half16_sum(rs);
#pragma unroll
      for (int nt = 0; nt < 4; nt++) o[nt][vi] *= alpha;
    }

    // Route P through wave-private LDS: C-layout -> A-layout
    unsigned short* pw = &Ps[wave][0][0];
#pragma unroll
    for (int nt = 0; nt < 4; nt++)
#pragma unroll
      for (int vi = 0; vi < 8; vi++)
        pw[(size_t)(vi + rhi) * 64 + nt * 16 + ln] = f2bf(s[nt][vi]);
    asm volatile("s_wait_dscnt 0" ::: "memory");  // same-wave LDS RAW fence

    bf16x16 pa0 = ld_afrag(&pw[(size_t)ln * 64], lane);        // keys 0..31
    bf16x16 pa1 = ld_afrag(&pw[(size_t)ln * 64 + 32], lane);   // keys 32..63
#pragma unroll
    for (int nt = 0; nt < 4; nt++) {  // output d-subtiles
      bf16x16 vb0 = ld_bfrag(&Vt[nt * 16 + ln][0], lane);
      bf16x16 vb1 = ld_bfrag(&Vt[nt * 16 + ln][32], lane);
      o[nt] = wmma_bf16(pa0, vb0, o[nt]);
      o[nt] = wmma_bf16(pa1, vb1, o[nt]);
    }
    __syncthreads();  // before next tile overwrites Ks/Vt
  };

  for (int kt = 0; kt < qt; ++kt) process_tile(kt, false);  // mask-free body
  process_tile(qt, true);                                   // peeled diagonal

  // Epilogue: normalize and store bf16 [B,S,H,D]
#pragma unroll
  for (int vi = 0; vi < 8; vi++) {
    float inv = 1.f / lrun[vi];
    int r = q0 + vi + rhi;
#pragma unroll
    for (int nt = 0; nt < 4; nt++) {
      int d = nt * 16 + ln;
      O[(((size_t)b * S_ + r) * H_ + h) * D_ + d] = f2bf(o[nt][vi] * inv);
    }
  }
}

// ---------- host orchestration ----------
extern "C" void kernel_launch(void* const* d_in, const int* in_sizes, int n_in,
                              void* d_out, int out_size, void* d_ws, size_t ws_size,
                              hipStream_t stream) {
  (void)in_sizes; (void)n_in; (void)out_size; (void)ws_size;
  const float* hs   = (const float*)d_in[0];
  const float* cosb = (const float*)d_in[1];
  const float* sinb = (const float*)d_in[2];
  // d_in[3] attention_mask: causal, reproduced analytically in-kernel
  const float* Wq = (const float*)d_in[4];
  const float* Wk = (const float*)d_in[5];
  const float* Wv = (const float*)d_in[6];
  const float* Wo = (const float*)d_in[7];

  char* ws = (char*)d_ws;
  size_t off = 0;
  auto alloc = [&](size_t bytes) {
    char* p = ws + off;
    off += (bytes + 255) & ~(size_t)255;
    return p;
  };
  unsigned short* Xb  = (unsigned short*)alloc((size_t)B_ * S_ * HID_ * 2);
  unsigned short* Wqb = (unsigned short*)alloc((size_t)H_ * D_ * HID_ * 2);
  unsigned short* Wkb = (unsigned short*)alloc((size_t)KVH_ * D_ * HID_ * 2);
  unsigned short* Wvb = (unsigned short*)alloc((size_t)KVH_ * D_ * HID_ * 2);
  unsigned short* Wob = (unsigned short*)alloc((size_t)HID_ * H_ * D_ * 2);
  unsigned short* Qb  = (unsigned short*)alloc((size_t)B_ * S_ * H_ * D_ * 2);
  unsigned short* Kb  = (unsigned short*)alloc((size_t)B_ * S_ * KVH_ * D_ * 2);
  unsigned short* Vb  = (unsigned short*)alloc((size_t)B_ * S_ * KVH_ * D_ * 2);
  unsigned short* Ab  = (unsigned short*)alloc((size_t)B_ * S_ * H_ * D_ * 2);

  auto cvt = [&](const float* src, unsigned short* dst, int n) {
    cvt_f32_bf16_kernel<<<(n + 255) / 256, 256, 0, stream>>>(src, dst, n);
  };
  cvt(hs, Xb, B_ * S_ * HID_);
  cvt(Wq, Wqb, H_ * D_ * HID_);
  cvt(Wk, Wkb, KVH_ * D_ * HID_);
  cvt(Wv, Wvb, KVH_ * D_ * HID_);
  cvt(Wo, Wob, HID_ * H_ * D_);

  const int M = B_ * S_;
  // Q projection pre-scaled by 1/sqrt(D); K/V unscaled (bf16 out, [b,s,h,d])
  gemm_bf16_kernel<false><<<dim3((H_ * D_) / 128, M / 128), 256, 0, stream>>>(
      Xb, Wqb, Qb, M, H_ * D_, HID_, 0.125f);
  gemm_bf16_kernel<false><<<dim3((KVH_ * D_) / 128, M / 128), 256, 0, stream>>>(
      Xb, Wkb, Kb, M, KVH_ * D_, HID_, 1.0f);
  gemm_bf16_kernel<false><<<dim3((KVH_ * D_) / 128, M / 128), 256, 0, stream>>>(
      Xb, Wvb, Vb, M, KVH_ * D_, HID_, 1.0f);

  // Partial RoPE in place
  rope_kernel<<<(B_ * S_ * H_ * 16 + 255) / 256, 256, 0, stream>>>(
      Qb, cosb, sinb, H_, B_ * S_ * H_ * 16);
  rope_kernel<<<(B_ * S_ * KVH_ * 16 + 255) / 256, 256, 0, stream>>>(
      Kb, cosb, sinb, KVH_, B_ * S_ * KVH_ * 16);

  // Causal GQA flash attention (TDM-staged K tiles)
  flash_kernel<<<dim3(S_ / 64, H_, B_), 128, 0, stream>>>(Qb, Kb, Vb, Ab);

  // Output projection -> f32
  gemm_bf16_kernel<true><<<dim3(HID_ / 128, M / 128), 256, 0, stream>>>(
      Ab, Wob, d_out, M, HID_, H_ * D_, 1.0f);
}
